// HistoLoss_76802605187293
// MI455X (gfx1250) — compile-verified, matrix operand order checked by reading, not run on hardware
//
#include <hip/hip_runtime.h>

typedef __attribute__((ext_vector_type(16))) _Float16 v16h;
typedef __attribute__((ext_vector_type(8)))  float    v8f;

#define B_N    1024
#define T_N    256
#define C_N    16
#define NB_N   64
#define BCHUNK 32
#define XS_STRIDE 36   // 16 channels * 36 floats, padded to dodge LDS bank conflicts
#define LOSS_SCALE (1.0f / (float)(T_N * C_N * NB_N))   // REG=1.0 folded in

// One block per t. 512 threads = 16 waves; wave w owns channel c=w.
// Per 32-wide b-chunk each wave packs 0/1 indicators (f16) into four 16x32 A
// fragments (one per 16-bin group) and reduces over b with
// v_wmma_f32_16x16x32_f16 against an all-ones B matrix. Counts accumulate
// exactly in the f32 C/D fragment across all 32 chunks.
__global__ __launch_bounds__(512) void histoloss_wmma_kernel(
    const float* __restrict__ x_fake,     // [B,T,C]
    const float* __restrict__ locs,       // [T,C,NB]
    const float* __restrict__ deltas,     // [T,C]
    const float* __restrict__ densities,  // [T,C,NB]
    float* __restrict__ partial)          // [T]
{
  __shared__ float xs[C_N * XS_STRIDE];
  __shared__ float wavesum[16];

  const int t    = blockIdx.x;
  const int tid  = threadIdx.x;
  const int c    = tid >> 5;        // wave id == channel
  const int lane = tid & 31;
  const int half = lane >> 4;       // 0: lanes 0-15, 1: lanes 16-31
  const int row  = lane & 15;       // A/D row within the 16-bin group

  const float delta      = deltas[t * C_N + c];
  const float hd         = 0.5f * delta;
  const float inv_Bdelta = 1.0f / ((float)B_N * delta);

  // lane (and lane+16) own bin m = g*16 + row for each of the 4 groups
  float loc[4];
  #pragma unroll
  for (int g = 0; g < 4; ++g)
    loc[g] = locs[(t * C_N + c) * NB_N + g * 16 + row];

  v16h ones;
  #pragma unroll
  for (int i = 0; i < 16; ++i) ones[i] = (_Float16)1.0f;

  v8f acc[4];
  #pragma unroll
  for (int g = 0; g < 4; ++g)
    #pragma unroll
    for (int i = 0; i < 8; ++i) acc[g][i] = 0.0f;

  const int bl = tid >> 4;   // chunk-local b this thread stages (0..31)
  const int cc = tid & 15;   // channel it stages (coalesced 16-float rows)

  for (int b0 = 0; b0 < B_N; b0 += BCHUNK) {
    // Stage x_fake[b0+bl, t, cc] -> xs[cc][bl]  (512 threads, 512 elements)
    xs[cc * XS_STRIDE + bl] =
        x_fake[(size_t)(b0 + bl) * (T_N * C_N) + t * C_N + cc];
    if (b0 + BCHUNK < B_N)  // global_prefetch_b8 for next chunk
      __builtin_prefetch(
          &x_fake[(size_t)(b0 + BCHUNK + bl) * (T_N * C_N) + t * C_N + cc], 0, 0);
    __syncthreads();

    // Each half-wave reads its 16 b-samples (broadcast within half-wave).
    float xv[16];
    #pragma unroll
    for (int i = 0; i < 16; ++i)
      xv[i] = xs[c * XS_STRIDE + half * 16 + i];

    #pragma unroll
    for (int g = 0; g < 4; ++g) {
      // A fragment: row = bin, K = b. With ones-B, any K permutation sums the
      // same, so we need not match the ISA K striping exactly.
      v16h a;
      #pragma unroll
      for (int i = 0; i < 16; ++i) {
        const float d = __builtin_fabsf(xv[i] - loc[g]);
        a[i] = (d < hd) ? (_Float16)1.0f : (_Float16)0.0f;
      }
      acc[g] = __builtin_amdgcn_wmma_f32_16x16x32_f16(
          /*neg_a=*/false, a, /*neg_b=*/false, ones,
          /*c_mod=*/(short)0, acc[g], /*reuse_a=*/false, /*reuse_b=*/false);
    }
    __syncthreads();
  }

  // D layout: VGPR r holds row r (lanes 0-15) / row r+8 (lanes 16-31); all
  // columns equal. Lanes 0 and 16 between them see all 16 rows of each group.
  float lsum = 0.0f;
  if (row == 0) {
    #pragma unroll
    for (int g = 0; g < 4; ++g) {
      #pragma unroll
      for (int r = 0; r < 8; ++r) {
        const int m = g * 16 + 8 * half + r;
        const float density = acc[g][r] * inv_Bdelta;
        const float dref = densities[(t * C_N + c) * NB_N + m];
        lsum += __builtin_fabsf(density - dref);
      }
    }
  }
  lsum += __shfl_down(lsum, 16, 32);           // lane0 += lane16
  if (lane == 0) wavesum[c] = lsum;
  __syncthreads();

  if (tid < 16) {
    float s = wavesum[tid];
    s += __shfl_down(s, 8, 32);
    s += __shfl_down(s, 4, 32);
    s += __shfl_down(s, 2, 32);
    s += __shfl_down(s, 1, 32);
    if (tid == 0) partial[t] = s * LOSS_SCALE;
  }
}

// Deterministic fixed-order tree reduction of the 256 per-t partials.
__global__ __launch_bounds__(256) void histoloss_reduce_kernel(
    const float* __restrict__ partial, float* __restrict__ out)
{
  __shared__ float s[T_N];
  const int tid = threadIdx.x;
  s[tid] = partial[tid];
  __syncthreads();
  for (int off = 128; off > 0; off >>= 1) {
    if (tid < off) s[tid] += s[tid + off];
    __syncthreads();
  }
  if (tid == 0) out[0] = s[0];
}

extern "C" void kernel_launch(void* const* d_in, const int* in_sizes, int n_in,
                              void* d_out, int out_size, void* d_ws, size_t ws_size,
                              hipStream_t stream) {
  (void)in_sizes; (void)n_in; (void)out_size; (void)ws_size;
  const float* x_fake    = (const float*)d_in[0];
  const float* locs      = (const float*)d_in[1];
  const float* deltas    = (const float*)d_in[2];
  const float* densities = (const float*)d_in[3];
  float* partial = (float*)d_ws;   // T_N floats of scratch, rewritten every call

  histoloss_wmma_kernel<<<T_N, 512, 0, stream>>>(x_fake, locs, deltas,
                                                 densities, partial);
  histoloss_reduce_kernel<<<1, T_N, 0, stream>>>(partial, (float*)d_out);
}
